// TransformerBlock_1159641170404
// MI455X (gfx1250) — matrix operand order repaired and checked by reading, not compiled, not run on hardware
//
#include <hip/hip_runtime.h>
#include <stdint.h>
#include <stddef.h>

typedef _Float16 v16h __attribute__((ext_vector_type(16)));
typedef float    v8f  __attribute__((ext_vector_type(8)));
typedef int      i32x4 __attribute__((__vector_size__(16)));

#define BB 2
#define SS 2048
#define DD 1024
#define HH 16
#define MM (BB*SS)      // 4096 rows
#define ASTR 40         // LDS stride (halves) for A tiles: 80B, 16B-aligned rows
#define BSTR 34         // LDS stride (halves) for transposed B tiles
#define KSTR 72         // LDS stride (halves) for attention K tiles: 144B, 16B-aligned
#define TMM 256         // GEMM block tile M
#define TNN 128         // GEMM block tile N

#if defined(__AMDGCN__) && __has_builtin(__builtin_amdgcn_global_load_async_to_lds_b128)
#define HAVE_ASYNC 1
#else
#define HAVE_ASYNC 0
#endif

#define ASG __attribute__((address_space(1)))
#define ASL __attribute__((address_space(3)))

union HFrag { v16h h; uint32_t u[8]; };

#if HAVE_ASYNC
__device__ __forceinline__ void async_copy16(const void* g, void* l) {
    __builtin_amdgcn_global_load_async_to_lds_b128(
        (ASG i32x4*)(i32x4*)(void*)g, (ASL i32x4*)(i32x4*)l, 0, 0);
}
#endif

__device__ __forceinline__ void wait_async0() {
#if defined(__AMDGCN__)
#if __has_builtin(__builtin_amdgcn_s_wait_asynccnt)
    __builtin_amdgcn_s_wait_asynccnt(0);
#else
    __asm__ volatile("s_wait_asynccnt 0" ::: "memory");
#endif
#endif
}

__device__ __forceinline__ float gelu_f(float x) {
    const float c = 0.7978845608028654f; // sqrt(2/pi)
    float t = tanhf(c * (x + 0.044715f * x * x * x));
    return 0.5f * x * (1.0f + t);
}

// ---------------- LayerNorm: f32 in -> f16 out -------------------------------
__global__ __launch_bounds__(256) void ln_kernel(const float* __restrict__ x,
                                                 const float* __restrict__ g,
                                                 const float* __restrict__ sb,
                                                 _Float16* __restrict__ out) {
    int row = blockIdx.x, tid = threadIdx.x;
    const float* xr = x + (size_t)row * DD;
    float sum = 0.f, sq = 0.f;
    for (int i = tid; i < DD; i += 256) { float v = xr[i]; sum += v; sq += v * v; }
    for (int off = 1; off < 32; off <<= 1) {
        sum += __shfl_xor(sum, off, 32);
        sq  += __shfl_xor(sq,  off, 32);
    }
    __shared__ float rs[8], rq[8];
    int wave = tid >> 5;
    if ((tid & 31) == 0) { rs[wave] = sum; rq[wave] = sq; }
    __syncthreads();
    float ts = 0.f, tq = 0.f;
#pragma unroll
    for (int w = 0; w < 8; ++w) { ts += rs[w]; tq += rq[w]; }
    float mean = ts * (1.0f / DD);
    float inv  = rsqrtf(tq * (1.0f / DD) - mean * mean + 1e-5f);
    _Float16* orow = out + (size_t)row * DD;
    for (int i = tid; i < DD; i += 256)
        orow[i] = (_Float16)((xr[i] - mean) * inv * g[i] + sb[i]);
}

// ---------------- Generic WMMA GEMM: C = act(A_f16 @ W_f32 + bias) + resid ---
// 256x128 tile per 256-thread block (8 waves); wave = 64x64 subtile = 4x4 WMMAs.
template <bool OUT_F16, bool GELU>
__global__ __launch_bounds__(256) void gemm_f16(const _Float16* __restrict__ A,
                                                const float* __restrict__ W,
                                                const float* __restrict__ bias,
                                                const float* __restrict__ resid,
                                                void* __restrict__ Cout,
                                                int M, int N, int K) {
    __shared__ _Float16 As[TMM * ASTR];   // As[m][k], row-major, 16B-aligned rows
    __shared__ _Float16 Bs[TNN * BSTR];   // Bs[n][k], transposed W tile
    int tid = threadIdx.x;
    int wave = tid >> 5, lane = tid & 31;
    int lh = lane >> 4, ll = lane & 15;
    int m0 = blockIdx.y * TMM, n0 = blockIdx.x * TNN;
    int wm = (wave >> 1) * 64;   // wave row offset inside tile (4 wave-rows)
    int wn = (wave & 1) * 64;    // wave col offset inside tile (2 wave-cols)

    v8f acc[4][4];
#pragma unroll
    for (int a = 0; a < 4; ++a)
#pragma unroll
        for (int b = 0; b < 4; ++b) acc[a][b] = v8f{};

    for (int k0 = 0; k0 < K; k0 += 32) {
        __syncthreads();
        // ---- stage A tile TMMx32 (f16 raw copy) ----
#if HAVE_ASYNC
        for (int i = tid; i < TMM * 4; i += 256) {      // 4 x 16B chunks per row
            int m = i >> 2, c = i & 3;
            async_copy16(A + (size_t)(m0 + m) * K + k0 + c * 8,
                         As + m * ASTR + c * 8);
        }
#else
        for (int i = tid; i < TMM * 32; i += 256) {
            int m = i >> 5, k = i & 31;
            As[m * ASTR + k] = A[(size_t)(m0 + m) * K + k0 + k];
        }
#endif
        // ---- stage W tile 32xTNN, transposed + f32->f16 (needs VALU convert) ----
        for (int i = tid; i < 32 * TNN; i += 256) {
            int k = i >> 7, n = i & 127;
            Bs[n * BSTR + k] = (_Float16)W[(size_t)(k0 + k) * N + n0 + n];
        }
        // prefetch next W tile into L2 while we compute
        if (k0 + 32 < K) {
            const float* nw = W + (size_t)(k0 + 32 + (tid >> 3)) * N + n0 + (tid & 7) * 16;
            __builtin_prefetch(nw, 0, 1);
        }
#if HAVE_ASYNC
        wait_async0();
#endif
        __syncthreads();

        HFrag af[4];
#pragma unroll
        for (int mt = 0; mt < 4; ++mt) {
            const uint32_t* rp = (const uint32_t*)(As + (wm + mt * 16 + ll) * ASTR);
#pragma unroll
            for (int j = 0; j < 4; ++j) {
                af[mt].u[j]     = rp[4 * lh + j];
                af[mt].u[4 + j] = rp[8 + 4 * lh + j];
            }
        }
#pragma unroll
        for (int nt = 0; nt < 4; ++nt) {
            HFrag bf;
            const uint32_t* cp = (const uint32_t*)(Bs + (wn + nt * 16 + ll) * BSTR);
#pragma unroll
            for (int j = 0; j < 4; ++j) { bf.u[j] = cp[4 * lh + j]; bf.u[4 + j] = cp[8 + 4 * lh + j]; }
#pragma unroll
            for (int mt = 0; mt < 4; ++mt)
                acc[mt][nt] = __builtin_amdgcn_wmma_f32_16x16x32_f16(
                    false, af[mt].h, false, bf.h, (short)0, acc[mt][nt], false, false);
        }
    }

#pragma unroll
    for (int mt = 0; mt < 4; ++mt)
#pragma unroll
        for (int nt = 0; nt < 4; ++nt)
#pragma unroll
            for (int r = 0; r < 8; ++r) {
                int row = m0 + wm + mt * 16 + r + 8 * lh;
                int col = n0 + wn + nt * 16 + ll;
                float v = acc[mt][nt][r];
                if (bias)  v += bias[col];
                if (GELU)  v  = gelu_f(v);
                if (resid) v += resid[(size_t)row * N + col];
                if (OUT_F16) ((_Float16*)Cout)[(size_t)row * N + col] = (_Float16)v;
                else         ((float*)Cout)[(size_t)row * N + col]    = v;
            }
}

// ---------------- Causal flash attention (per b, head, 64-query block) -------
// 128 threads = 4 waves; wave w owns 16 query rows. KV processed 32 keys/step.
__global__ __launch_bounds__(128) void attn_kernel(const _Float16* __restrict__ Q,
                                                   const _Float16* __restrict__ Kg,
                                                   const _Float16* __restrict__ Vg,
                                                   _Float16* __restrict__ ctx) {
    int qblk = blockIdx.x, h = blockIdx.y, b = blockIdx.z;
    int tid = threadIdx.x, wave = tid >> 5, lane = tid & 31;
    int lh = lane >> 4, ll = lane & 15;
    int qrow = qblk * 64 + wave * 16;
    size_t base = (size_t)b * SS * DD + (size_t)h * 64;  // [b, :, h*64 + :]

    __shared__ _Float16 Ks[32 * KSTR];   // Ks[j][d], 16B-aligned rows
    __shared__ _Float16 Vt[64 * BSTR];   // Vt[d][j] (transposed V)
    __shared__ _Float16 Ps[4][16 * BSTR];

    // resident Q A-fragments (d-chunks 0..31, 32..63)
    HFrag qf[2];
#pragma unroll
    for (int dt = 0; dt < 2; ++dt) {
        const uint32_t* qp =
            (const uint32_t*)(Q + base + (size_t)(qrow + ll) * DD + dt * 32);
#pragma unroll
        for (int j = 0; j < 4; ++j) {
            qf[dt].u[j]     = qp[4 * lh + j];
            qf[dt].u[4 + j] = qp[8 + 4 * lh + j];
        }
    }

    v8f o[4];
#pragma unroll
    for (int dt = 0; dt < 4; ++dt) o[dt] = v8f{};
    float mrow[8], lrow[8];
#pragma unroll
    for (int r = 0; r < 8; ++r) { mrow[r] = -1e30f; lrow[r] = 0.f; }

    int kend = qblk * 64 + 64;
    for (int k0 = 0; k0 < kend; k0 += 32) {
        __syncthreads();
        // ---- stage K block (raw f16 copy, async path) ----
#if HAVE_ASYNC
        for (int i = tid; i < 32 * 8; i += 128) {       // 8 x 16B chunks per key row
            int j = i >> 3, c = i & 7;
            async_copy16(Kg + base + (size_t)(k0 + j) * DD + c * 8,
                         Ks + j * KSTR + c * 8);
        }
#else
        for (int i = tid; i < 32 * 64; i += 128) {
            int j = i >> 6, d = i & 63;
            Ks[j * KSTR + d] = Kg[base + (size_t)(k0 + j) * DD + d];
        }
#endif
        // ---- stage V block transposed (needs shuffle, manual) ----
        for (int i = tid; i < 32 * 64; i += 128) {
            int j = i >> 6, d = i & 63;
            Vt[d * BSTR + j] = Vg[base + (size_t)(k0 + j) * DD + d];
        }
#if HAVE_ASYNC
        wait_async0();
#endif
        __syncthreads();

        // scores: 16 q-rows x 32 keys, contraction over HD=64
        v8f sc[2];
        sc[0] = v8f{}; sc[1] = v8f{};
#pragma unroll
        for (int nt = 0; nt < 2; ++nt)
#pragma unroll
            for (int dt = 0; dt < 2; ++dt) {
                HFrag kf;
                const uint32_t* kp =
                    (const uint32_t*)(Ks + (nt * 16 + ll) * KSTR + dt * 32);
#pragma unroll
                for (int j = 0; j < 4; ++j) { kf.u[j] = kp[4 * lh + j]; kf.u[4 + j] = kp[8 + 4 * lh + j]; }
                sc[nt] = __builtin_amdgcn_wmma_f32_16x16x32_f16(
                    false, qf[dt].h, false, kf.h, (short)0, sc[nt], false, false);
            }

        // scale + causal mask
#pragma unroll
        for (int nt = 0; nt < 2; ++nt)
#pragma unroll
            for (int r = 0; r < 8; ++r) {
                int q = qrow + r + 8 * lh;
                int k = k0 + nt * 16 + ll;
                float v = sc[nt][r] * 0.125f;  // 1/sqrt(64)
                sc[nt][r] = (k > q) ? -1e30f : v;
            }

        // online softmax per row (row spread across a 16-lane group)
#pragma unroll
        for (int r = 0; r < 8; ++r) {
            float mx = fmaxf(sc[0][r], sc[1][r]);
            for (int off = 1; off < 16; off <<= 1) mx = fmaxf(mx, __shfl_xor(mx, off, 32));
            float mnew = fmaxf(mrow[r], mx);
            float p0 = __expf(sc[0][r] - mnew);
            float p1 = __expf(sc[1][r] - mnew);
            sc[0][r] = p0; sc[1][r] = p1;
            float s = p0 + p1;
            for (int off = 1; off < 16; off <<= 1) s += __shfl_xor(s, off, 32);
            float scale = __expf(mrow[r] - mnew);
            lrow[r] = lrow[r] * scale + s;
            mrow[r] = mnew;
#pragma unroll
            for (int dt = 0; dt < 4; ++dt) o[dt][r] *= scale;
        }

        // convert P: C-layout -> A-fragment via per-wave LDS round trip
        _Float16* pw = &Ps[wave][0];
#pragma unroll
        for (int nt = 0; nt < 2; ++nt)
#pragma unroll
            for (int r = 0; r < 8; ++r)
                pw[(r + 8 * lh) * BSTR + nt * 16 + ll] = (_Float16)sc[nt][r];
        __asm__ volatile("s_wait_dscnt 0" ::: "memory");
        HFrag pf;
        const uint32_t* pp = (const uint32_t*)(pw + ll * BSTR);
#pragma unroll
        for (int j = 0; j < 4; ++j) {
            pf.u[j]     = pp[4 * lh + j];
            pf.u[4 + j] = pp[8 + 4 * lh + j];
        }

        // O += P @ V  (4 output d-tiles, K=32)
#pragma unroll
        for (int dt = 0; dt < 4; ++dt) {
            HFrag vf;
            const uint32_t* vp = (const uint32_t*)(Vt + (dt * 16 + ll) * BSTR);
#pragma unroll
            for (int j = 0; j < 4; ++j) { vf.u[j] = vp[4 * lh + j]; vf.u[4 + j] = vp[8 + 4 * lh + j]; }
            o[dt] = __builtin_amdgcn_wmma_f32_16x16x32_f16(
                false, pf.h, false, vf.h, (short)0, o[dt], false, false);
        }
    }

    // normalize and store ctx (f16, [b, q, h*64+d] layout)
#pragma unroll
    for (int dt = 0; dt < 4; ++dt)
#pragma unroll
        for (int r = 0; r < 8; ++r) {
            int q = qrow + r + 8 * lh;
            int col = dt * 16 + ll;
            ctx[base + (size_t)q * DD + col] = (_Float16)(o[dt][r] / lrow[r]);
        }
}

// ---------------- Host-side pipeline -----------------------------------------
extern "C" void kernel_launch(void* const* d_in, const int* in_sizes, int n_in,
                              void* d_out, int out_size, void* d_ws, size_t ws_size,
                              hipStream_t stream) {
    (void)in_sizes; (void)n_in; (void)out_size; (void)ws_size;
    const float* x  = (const float*)d_in[0];
    const float* Wq = (const float*)d_in[1];
    const float* Wk = (const float*)d_in[2];
    const float* Wv = (const float*)d_in[3];
    const float* Wo = (const float*)d_in[4];
    const float* bo = (const float*)d_in[5];
    const float* W1 = (const float*)d_in[6];
    const float* b1 = (const float*)d_in[7];
    const float* W2 = (const float*)d_in[8];
    const float* b2 = (const float*)d_in[9];
    const float* g1 = (const float*)d_in[10];
    const float* s1 = (const float*)d_in[11];
    const float* g2 = (const float*)d_in[12];
    const float* s2 = (const float*)d_in[13];

    char* ws = (char*)d_ws;
    _Float16* h16  = (_Float16*)(ws);                         // 8 MB
    _Float16* Q16  = (_Float16*)(ws + ((size_t)8  << 20));    // 8 MB
    _Float16* K16  = (_Float16*)(ws + ((size_t)16 << 20));    // 8 MB
    _Float16* V16  = (_Float16*)(ws + ((size_t)24 << 20));    // 8 MB
    _Float16* C16  = (_Float16*)(ws + ((size_t)32 << 20));    // 8 MB
    float*    x1   = (float*)   (ws + ((size_t)40 << 20));    // 16 MB
    _Float16* mid  = (_Float16*)(ws + ((size_t)56 << 20));    // 32 MB
    float*    out  = (float*)d_out;

    dim3 gD(DD / TNN, MM / TMM);          // N=1024 GEMMs: (8, 16)
    dim3 g4(4 * DD / TNN, MM / TMM);      // N=4096 GEMM:  (32, 16)

    // LN1
    ln_kernel<<<MM, 256, 0, stream>>>(x, g1, s1, h16);
    // QKV projections (f16 out)
    gemm_f16<true,  false><<<gD, 256, 0, stream>>>(h16, Wq, nullptr, nullptr, Q16, MM, DD, DD);
    gemm_f16<true,  false><<<gD, 256, 0, stream>>>(h16, Wk, nullptr, nullptr, K16, MM, DD, DD);
    gemm_f16<true,  false><<<gD, 256, 0, stream>>>(h16, Wv, nullptr, nullptr, V16, MM, DD, DD);
    // causal attention
    attn_kernel<<<dim3(SS / 64, HH, BB), 128, 0, stream>>>(Q16, K16, V16, C16);
    // output projection + bias + residual (f32 out)
    gemm_f16<false, false><<<gD, 256, 0, stream>>>(C16, Wo, bo, x, x1, MM, DD, DD);
    // LN2
    ln_kernel<<<MM, 256, 0, stream>>>(x1, g2, s2, h16);
    // MLP up + GELU (f16 out)
    gemm_f16<true,  true ><<<g4, 256, 0, stream>>>(h16, W1, b1, nullptr, mid, MM, 4 * DD, DD);
    // MLP down + bias + residual -> final output (f32)
    gemm_f16<false, false><<<gD, 256, 0, stream>>>(mid, W2, b2, x1, out, MM, DD, 4 * DD);
}
